// NewMoE_62225486184915
// MI455X (gfx1250) — matrix operand order repaired and checked by reading, not compiled, not run on hardware
//
#include <hip/hip_runtime.h>

typedef _Float16 half_t;
typedef __attribute__((ext_vector_type(16))) _Float16 v16h;
typedef __attribute__((ext_vector_type(8)))  _Float16 v8h;
typedef __attribute__((ext_vector_type(8)))  float    v8f;
typedef __attribute__((ext_vector_type(4)))  float    v4f;

#define NBATCH 8192
#define NG 3
#define NE 32
#define ND 128
#define CAP 4096                 // max entries per expert bucket (random ~768 expected)
#define MAXTILES (31 * (CAP / 16))
#define DUMP_ROW (NBATCH * NG)   // garbage row for padded tile rows

// ---------------------------------------------------------------------------
// init: zero the per-(token,genre) accumulator slots + expert counts, and
// convert W0/W1 to f16 transposed layout WT[e][out][in] so each WMMA
// B-fragment is one contiguous 32B load.
// ---------------------------------------------------------------------------
__global__ void moe_init_kernel(const float* __restrict__ W0,
                                const float* __restrict__ W1,
                                half_t* __restrict__ W0T,
                                half_t* __restrict__ W1T,
                                float* __restrict__ acc3,
                                int* __restrict__ counts) {
    int idx = blockIdx.x * blockDim.x + threadIdx.x;
    if (idx < NBATCH * NG * ND) acc3[idx] = 0.0f;
    if (idx < NE * ND * ND) {
        int e = idx / (ND * ND);
        int r = idx % (ND * ND);
        int i = r / ND;          // in  (K)
        int o = r % ND;          // out (N)
        W0T[e * ND * ND + o * ND + i] = (half_t)W0[idx];
        W1T[e * ND * ND + o * ND + i] = (half_t)W1[idx];
    }
    if (idx < NE) counts[idx] = 0;
}

// ---------------------------------------------------------------------------
// bucket: scatter (token,genre-slot) pairs into per-expert lists, packed as
// (b<<2)|g so every bucket entry owns a unique output slot (no atomics in
// the GEMM epilogue). Genre value 0 is the masked padding slot.
// ---------------------------------------------------------------------------
__global__ void moe_bucket_kernel(const int* __restrict__ genres,
                                  int* __restrict__ counts,
                                  int* __restrict__ buckets) {
    int idx = blockIdx.x * blockDim.x + threadIdx.x;
    if (idx >= NBATCH * NG) return;
    int b = idx / NG;
    int g = idx % NG;
    int e = genres[idx];
    if (e > 0 && e < NE) {
        int pos = atomicAdd(&counts[e], 1);
        if (pos < CAP) buckets[e * CAP + pos] = (b << 2) | g;
    }
}

// ---------------------------------------------------------------------------
// tiledir: build flat list of (expert, row-base) 16-row M-tiles.
// ---------------------------------------------------------------------------
__global__ void moe_tiledir_kernel(const int* __restrict__ counts,
                                   int* __restrict__ ntiles,
                                   int2* __restrict__ tiles) {
    __shared__ int tc[NE];
    int e = threadIdx.x;
    int c = (e > 0) ? min(counts[e], CAP) : 0;
    int t = (c + 15) >> 4;
    tc[e] = t;
    __syncthreads();
    if (e == 0) {
        int sum = 0;
        for (int i = 0; i < NE; i++) { int v = tc[i]; tc[i] = sum; sum += v; }
        *ntiles = sum;
    }
    __syncthreads();
    int base = tc[e];
    for (int i = 0; i < t; i++) tiles[base + i] = make_int2(e, i * 16);
}

// ---------------------------------------------------------------------------
// persistent expert-GEMM: 8 waves per WG, each wave owns a 16-wide N-slice;
// K=128 consumed by 4x v_wmma_f32_16x16x32_f16. Branch-free epilogue: plain
// stores into per-(token,genre) rows; padded rows target a dump row.
// F32A selects f32 source (layer 1: x) vs f16 source (layer 2: h16).
// ---------------------------------------------------------------------------
template <bool F32A>
__global__ void __launch_bounds__(256)
moe_gemm_kernel(const void* __restrict__ Asrc,
                const half_t* __restrict__ WT,
                const int* __restrict__ counts,
                const int* __restrict__ buckets,
                const int* __restrict__ ntiles_p,
                const int2* __restrict__ tiles,
                float* __restrict__ acc3) {
    __shared__ half_t xs[16 * ND];   // 16-row A tile, row major f16
    __shared__ int    toks[16];      // token id per row, -1 = pad
    __shared__ int    rowoff[16];    // acc3 row (b*NG+g), DUMP_ROW for pads

    const int ntiles = *ntiles_p;
    const int lane   = threadIdx.x & 31;
    const int wave   = threadIdx.x >> 5;
    const int l15    = lane & 15;    // N within slice
    const int khalf  = lane >> 4;    // K-half selector (contiguous K order chosen here)
    const int n0     = wave * 16;
    const int trow   = threadIdx.x >> 4;   // staging: row 0..15
    const int tseg   = threadIdx.x & 15;   // staging: 8-col segment 0..15

    for (int t = blockIdx.x; t < ntiles; t += gridDim.x) {
        const int2 td  = tiles[t];
        const int e    = td.x;
        const int base = td.y;
        const int cnt  = min(counts[e], CAP);

        if (threadIdx.x < 16) {
            int p = base + (int)threadIdx.x;
            int s = (p < cnt) ? buckets[e * CAP + p] : -1;
            toks[threadIdx.x]   = (s >= 0) ? (s >> 2) : -1;
            rowoff[threadIdx.x] = (s >= 0) ? ((s >> 2) * NG + (s & 3)) : DUMP_ROW;
        }
        __syncthreads();

        // Stage A: one 8-element row segment per thread (vectorized).
        {
            int tok = toks[trow];
            v8h hv = {};
            if (tok >= 0) {
                if (F32A) {
                    const float* src = (const float*)Asrc + tok * ND + tseg * 8;
                    v4f f0 = *(const v4f*)(src);
                    v4f f1 = *(const v4f*)(src + 4);
                    #pragma unroll
                    for (int j = 0; j < 4; j++) {
                        hv[j]     = (half_t)f0[j];
                        hv[j + 4] = (half_t)f1[j];
                    }
                } else {
                    hv = *(const v8h*)((const half_t*)Asrc + tok * ND + tseg * 8);
                }
            }
            *(v8h*)&xs[trow * ND + tseg * 8] = hv;
        }
        __syncthreads();

        // Hoist this lane's 8 output row offsets (unconditional stores later).
        int myoff[8];
        #pragma unroll
        for (int r = 0; r < 8; r++) myoff[r] = rowoff[r + 8 * khalf];

        const half_t* We = WT + e * ND * ND;
        v8f c = {};
        #pragma unroll
        for (int kc = 0; kc < 4; kc++) {
            // A 16x32 f16 layout: lane l15 = row M; halves[0..15]=K kA..kA+15,
            // with kA = kc*32 + khalf*16 (matches B's contiguous K order).
            int kA = kc * 32 + khalf * 16;
            v8h a_lo = *(const v8h*)&xs[l15 * ND + kA];
            v8h a_hi = *(const v8h*)&xs[l15 * ND + kA + 8];
            v16h a;
            #pragma unroll
            for (int j = 0; j < 8; j++) { a[j] = a_lo[j]; a[8 + j] = a_hi[j]; }
            // B 32x16 f16 layout: lane l15 = column N, lane group = K half;
            // 16 contiguous K halves from transposed weights (32B load).
            int kB = kc * 32 + khalf * 16;
            v16h bf = *(const v16h*)&We[(n0 + l15) * ND + kB];
            c = __builtin_amdgcn_wmma_f32_16x16x32_f16(
                    false, a, false, bf, (short)0, c, false, false);
        }

        // C/D layout: lanes 0-15 -> M=r, lanes 16-31 -> M=r+8; N = l15.
        // Branch-free: pad rows write to DUMP_ROW (reserved garbage row).
        #pragma unroll
        for (int r = 0; r < 8; r++) {
            acc3[myoff[r] * ND + n0 + l15] = c[r];
        }
        __syncthreads();
    }
}

// ---------------------------------------------------------------------------
// finalize: sum masked genre slots, average, leaky ReLU; intermediate layer
// emits f16 activations for the next GEMM.
// ---------------------------------------------------------------------------
template <bool LAST>
__global__ void moe_finalize_kernel(const int* __restrict__ genres,
                                    const float* __restrict__ acc3,
                                    half_t* __restrict__ h16,
                                    float* __restrict__ out) {
    int idx = blockIdx.x * blockDim.x + threadIdx.x;
    if (idx >= NBATCH * ND) return;
    int b = idx >> 7;
    int d = idx & (ND - 1);
    float s = 0.0f;
    int cnt = 0;
    #pragma unroll
    for (int g = 0; g < NG; g++) {
        if (genres[b * NG + g] != 0) {
            cnt++;
            s += acc3[(b * NG + g) * ND + d];
        }
    }
    float v = (cnt > 0) ? (s / (float)cnt) : 0.0f;
    v = (v > 0.0f) ? v : 0.01f * v;
    if (LAST) out[idx] = v;
    else      h16[idx] = (half_t)v;
}

// ---------------------------------------------------------------------------
extern "C" void kernel_launch(void* const* d_in, const int* in_sizes, int n_in,
                              void* d_out, int out_size, void* d_ws, size_t ws_size,
                              hipStream_t stream) {
    const float* x      = (const float*)d_in[0];
    const int*   genres = (const int*)d_in[1];
    const float* W0     = (const float*)d_in[2];
    const float* W1     = (const float*)d_in[3];
    float*       out    = (float*)d_out;

    char* ws = (char*)d_ws;
    float*  acc3    = (float*)ws;                    // 12 MB + dump row: [B*3+1][128]
    half_t* h16     = (half_t*)(ws + (13u << 20));   // 2 MB
    half_t* W0T     = (half_t*)(ws + (15u << 20));   // 1 MB
    half_t* W1T     = (half_t*)(ws + (16u << 20));   // 1 MB
    int*    counts  = (int*)(ws + (17u << 20));      // 32 ints
    int*    ntiles  = counts + NE;
    int2*   tiles   = (int2*)(ws + (17u << 20) + 384);  // 8B aligned
    int*    buckets = (int*)((char*)tiles + (size_t)MAXTILES * sizeof(int2));

    moe_init_kernel<<<(NBATCH * NG * ND + 255) / 256, 256, 0, stream>>>(
        W0, W1, W0T, W1T, acc3, counts);
    moe_bucket_kernel<<<(NBATCH * NG + 255) / 256, 256, 0, stream>>>(
        genres, counts, buckets);
    moe_tiledir_kernel<<<1, NE, 0, stream>>>(counts, ntiles, tiles);

    // Layer 1: A from f32 x
    moe_gemm_kernel<true><<<256, 256, 0, stream>>>(
        x, W0T, counts, buckets, ntiles, tiles, acc3);
    moe_finalize_kernel<false><<<(NBATCH * ND + 255) / 256, 256, 0, stream>>>(
        genres, acc3, h16, out);

    // Layer 2: A from f16 hidden (slots fully overwritten, no re-zero needed)
    moe_gemm_kernel<false><<<256, 256, 0, stream>>>(
        h16, W1T, counts, buckets, ntiles, tiles, acc3);
    moe_finalize_kernel<true><<<(NBATCH * ND + 255) / 256, 256, 0, stream>>>(
        genres, acc3, h16, out);
}
